// RNN_MultiRegional_25855703122674
// MI455X (gfx1250) — matrix operand order repaired and checked by reading, not compiled, not run on hardware
//
#include <hip/hip_runtime.h>
#include <hip/hip_bf16.h>

#define T_STEPS   1000
#define BATCH     32
#define HID       1024
#define HALFD     512
#define EXC       359      /* HALF - N_INHIB = 512 - 153 */
#define NWG       16
#define NR        64       /* neurons per workgroup */
#define LDS_STRIDE 1032    /* 1024 + 8 bf16 pad to dodge bank conflicts */
#define TCONST    0.1f

/* output layout (floats, concatenated in reference return order) */
#define MEAN_OFF   0
#define STD_OFF    32000
#define RNN_OFF    64000
#define HN_OFF     32832000
#define XLAST_OFF  32864768
#define XOUT_OFF   32897536

typedef __attribute__((ext_vector_type(16))) __bf16 v16bf;
typedef __attribute__((ext_vector_type(8)))  __bf16 v8bf;
typedef __attribute__((ext_vector_type(8)))  float  v8f;

/* ---- init: reset h double-buffer (bf16 of hn) and barrier state ---- */
__global__ void init_kernel(const float* __restrict__ hn,
                            __bf16* __restrict__ hbuf,
                            unsigned* __restrict__ bar) {
  int i = blockIdx.x * blockDim.x + threadIdx.x;
  if (i < BATCH * HID) {
    hbuf[i] = (__bf16)hn[i];
    hbuf[BATCH * HID + i] = (__bf16)0.0f;
  }
  if (i == 0) { bar[0] = 0u; bar[1] = 0u; }
}

/* ---- persistent RNN scan: 16 blocks (one per WGP), 8 waves each ---- */
__global__ void __launch_bounds__(256, 1)
rnn_persistent(const float* __restrict__ inp,      /* [B,T,4]   */
               const float* __restrict__ x0,       /* [1,B,HID] */
               const float* __restrict__ w_str2m1, /* [512,512] */
               const float* __restrict__ w_m12m1,
               const float* __restrict__ w_m12str,
               const float* __restrict__ fixedw,
               const float* __restrict__ inp_w,    /* [4,HID]   */
               float* __restrict__ out,
               __bf16* __restrict__ hbuf,
               unsigned* __restrict__ bar) {
  extern __shared__ __align__(16) char smem[];
  __bf16* Ws = (__bf16*)smem;                            /* [NR][LDS_STRIDE]    */
  __bf16* Hs = (__bf16*)(smem + NR * LDS_STRIDE * 2);    /* [BATCH][LDS_STRIDE] */

  const int wg    = blockIdx.x;
  const int tid   = threadIdx.x;
  const int lane  = tid & 31;
  const int wave  = tid >> 5;
  const int rbase = wg * NR;

  /* Build W_rec slice (rows rbase..rbase+63) in bf16 directly into LDS.
     W_rec[r][c]: r<512: c<512 -> -fixed; c>=512 -> mask*relu(m12str)
                  r>=512: c<512 -> relu(str2m1); c>=512 -> +/-relu(m12m1)  */
  for (int k = tid; k < NR * HID; k += 256) {
    int row = k >> 10, c = k & 1023;
    int r = rbase + row;
    float v;
    if (r < HALFD) {
      if (c < HALFD) {
        v = -fixedw[r * HALFD + c];
      } else {
        int cc = c - HALFD;
        v = (cc < EXC) ? fmaxf(w_m12str[r * HALFD + cc], 0.0f) : 0.0f;
      }
    } else {
      int rr = r - HALFD;
      if (c < HALFD) {
        v = fmaxf(w_str2m1[rr * HALFD + c], 0.0f);
      } else {
        int cc = c - HALFD;
        float w = fmaxf(w_m12m1[rr * HALFD + cc], 0.0f);
        v = (cc < EXC) ? w : -w;
      }
    }
    Ws[row * LDS_STRIDE + c] = (__bf16)v;
  }

  /* tile assignment: wave -> (batch 16-tile, neuron 16-tile) */
  const int mtile = wave & 1;
  const int ntile = wave >> 1;
  const int nloc  = ntile * 16 + (lane & 15);   /* B/C/D column, local W row */
  const int nglob = rbase + nloc;
  const int mrow  = mtile * 16 + (lane & 15);   /* A-matrix row (batch)      */
  const int bbase = mtile * 16 + ((lane >= 16) ? 8 : 0); /* C rows: bbase+j  */

  /* preload input-projection weights for this neuron (str half only) */
  const bool has_inp = (nglob < HALFD);
  float wi0 = 0.f, wi1 = 0.f, wi2 = 0.f, wi3 = 0.f;
  if (has_inp) {
    wi0 = inp_w[0 * HID + nglob]; wi1 = inp_w[1 * HID + nglob];
    wi2 = inp_w[2 * HID + nglob]; wi3 = inp_w[3 * HID + nglob];
  }

  /* x state persists in the WMMA C/D accumulator layout */
  v8f xacc;
  for (int j = 0; j < 8; ++j) xacc[j] = x0[(bbase + j) * HID + nglob];

  float* rnn_out = out + RNN_OFF;
  float* x_out   = out + XOUT_OFF;

  volatile unsigned* cnt = bar;
  volatile unsigned* gen = bar + 1;

  const int aoff0 = (lane < 16) ? 0 : 8;    /* 16-bit A layout: split K per half-wave */
  const int aoff1 = (lane < 16) ? 16 : 24;
  const int boff  = (lane >= 16) ? 8 : 0;   /* 16-bit B layout: split K per half-wave */

  for (int t = 0; t < T_STEPS; ++t) {
    const __bf16* hsrc = hbuf + (size_t)(t & 1) * (BATCH * HID);
    __bf16*       hdst = hbuf + (size_t)((t + 1) & 1) * (BATCH * HID);

    __syncthreads();  /* previous step's LDS consumers done (covers Ws build at t=0) */

    /* stage h[t] (64 KB bf16) into padded LDS via async global->LDS copies:
       GLOBAL_LOAD_ASYNC_TO_LDS_B128 (GV mode), tracked by ASYNCcnt.        */
    for (int k = tid; k < (BATCH * HID) / 8; k += 256) {
      int row = k >> 7, ch = k & 127;
      unsigned lds_addr =
          (unsigned)(uintptr_t)(Hs + row * LDS_STRIDE + ch * 8);
      unsigned long long gaddr =
          (unsigned long long)(uintptr_t)(hsrc + row * HID + ch * 8);
      asm volatile("global_load_async_to_lds_b128 %0, %1, off"
                   :: "v"(lds_addr), "v"(gaddr) : "memory");
    }
    asm volatile("s_wait_asynccnt 0x0" ::: "memory");
    __syncthreads();

    /* 16x16 tile of h @ W_recT : 32 bf16 WMMAs over K=1024 */
    v8f acc = {};
    for (int kb = 0; kb < 32; ++kb) {
      int kbase = kb * 32;
      v8bf a0 = *(const v8bf*)(Hs + mrow * LDS_STRIDE + kbase + aoff0);
      v8bf a1 = *(const v8bf*)(Hs + mrow * LDS_STRIDE + kbase + aoff1);
      v8bf b0 = *(const v8bf*)(Ws + nloc * LDS_STRIDE + kbase + boff);
      v8bf b1 = *(const v8bf*)(Ws + nloc * LDS_STRIDE + kbase + boff + 16);
      v16bf A, Bm;
      for (int i = 0; i < 8; ++i) {
        A[i] = a0[i];  A[i + 8] = a1[i];
        Bm[i] = b0[i]; Bm[i + 8] = b1[i];
      }
      acc = __builtin_amdgcn_wmma_f32_16x16x32_bf16(
          false, A, false, Bm, (short)0, acc, false, false);
    }

    /* leaky integration + relu + stores */
    for (int j = 0; j < 8; ++j) {
      int b = bbase + j;
      float u = 0.0f;
      if (has_inp) {
        float4 iv = ((const float4*)inp)[b * T_STEPS + t];
        u = iv.x * wi0 + iv.y * wi1 + iv.z * wi2 + iv.w * wi3;
      }
      float xn = (1.0f - TCONST) * xacc[j] + TCONST * (acc[j] + u);
      xacc[j] = xn;
      float hv = fmaxf(xn, 0.0f);
      size_t idx = ((size_t)b * T_STEPS + t) * HID + nglob;
      rnn_out[idx] = hv;
      x_out[idx]   = xn;
      hdst[b * HID + nglob] = (__bf16)hv;
    }

    /* grid-wide sense-reversing barrier (16 co-resident blocks) */
    __threadfence();
    __syncthreads();
    if (tid == 0) {
      unsigned g = *gen;
      unsigned arrived = atomicAdd((unsigned*)cnt, 1u);
      if (arrived == NWG - 1) {
        *cnt = 0u;
        __threadfence();
        atomicAdd((unsigned*)gen, 1u);
      } else {
        while (*gen == g) { __builtin_amdgcn_s_sleep(1); }
      }
    }
    __syncthreads();
  }
}

/* ---- heads: mean/std = (rnn * alm_mask) @ W + b, one wave per (b,t) ---- */
__global__ void head_kernel(const float* __restrict__ rnn,
                            const float* __restrict__ mW, const float* __restrict__ mb,
                            const float* __restrict__ sW, const float* __restrict__ sb,
                            float* __restrict__ mean_out, float* __restrict__ std_out) {
  int wid  = (blockIdx.x * blockDim.x + threadIdx.x) >> 5;
  int lane = threadIdx.x & 31;
  if (wid >= BATCH * T_STEPS) return;
  int b = wid / T_STEPS, t = wid % T_STEPS;
  const float* row = rnn + ((size_t)b * T_STEPS + t) * HID + HALFD;
  float sm = 0.f, ss = 0.f;
  for (int k = lane; k < HALFD; k += 32) {
    float g = row[k];
    sm += g * mW[HALFD + k];
    ss += g * sW[HALFD + k];
  }
  for (int off = 16; off; off >>= 1) {
    sm += __shfl_xor(sm, off, 32);
    ss += __shfl_xor(ss, off, 32);
  }
  if (lane == 0) {
    mean_out[wid] = sm + mb[0];
    std_out[wid]  = ss + sb[0];
  }
}

/* ---- hn_last / x_last copies ---- */
__global__ void last_kernel(float* __restrict__ out) {
  int i = blockIdx.x * blockDim.x + threadIdx.x;
  if (i >= BATCH * HID) return;
  int b = i >> 10, h = i & 1023;
  size_t idx = ((size_t)b * T_STEPS + (T_STEPS - 1)) * HID + h;
  out[HN_OFF + i]    = out[RNN_OFF + idx];
  out[XLAST_OFF + i] = out[XOUT_OFF + idx];
}

extern "C" void kernel_launch(void* const* d_in, const int* in_sizes, int n_in,
                              void* d_out, int out_size, void* d_ws, size_t ws_size,
                              hipStream_t stream) {
  const float* inp       = (const float*)d_in[0];
  const float* hn        = (const float*)d_in[1];
  const float* x0        = (const float*)d_in[2];
  /* d_in[3] = w_str2str: multiplied by an all-zero mask in the reference */
  const float* w_str2m1  = (const float*)d_in[4];
  const float* w_m12m1   = (const float*)d_in[5];
  const float* w_m12str  = (const float*)d_in[6];
  const float* fixedw    = (const float*)d_in[7];
  const float* inp_w     = (const float*)d_in[8];
  const float* mW        = (const float*)d_in[9];
  const float* mb        = (const float*)d_in[10];
  const float* sW        = (const float*)d_in[11];
  const float* sb        = (const float*)d_in[12];
  float* out = (float*)d_out;

  __bf16*   hbuf = (__bf16*)d_ws;                                  /* 2 x 64 KB  */
  unsigned* bar  = (unsigned*)((char*)d_ws + 2 * BATCH * HID * 2); /* cnt, gen   */

  init_kernel<<<(BATCH * HID + 255) / 256, 256, 0, stream>>>(hn, hbuf, bar);

  size_t lds_bytes = (size_t)(NR + BATCH) * LDS_STRIDE * 2;  /* ~194 KB of 320 KB */
  rnn_persistent<<<NWG, 256, lds_bytes, stream>>>(
      inp, x0, w_str2m1, w_m12m1, w_m12str, fixedw, inp_w, out, hbuf, bar);

  head_kernel<<<(BATCH * T_STEPS) / 8, 256, 0, stream>>>(
      out + RNN_OFF, mW, mb, sW, sb, out + MEAN_OFF, out + STD_OFF);

  last_kernel<<<(BATCH * HID + 255) / 256, 256, 0, stream>>>(out);
}
